// Capsule_6004364280312
// MI455X (gfx1250) — matrix operand order repaired and checked by reading, not compiled, not run on hardware
//
#include <hip/hip_runtime.h>

// ---------------- types ----------------
typedef __attribute__((ext_vector_type(16))) __bf16    v16bf;
typedef __attribute__((ext_vector_type(8)))  __bf16    v8bf;
typedef __attribute__((ext_vector_type(8)))  float     v8f;
typedef __attribute__((ext_vector_type(8)))  _Float16  v8h;

// ---------------- problem sizes ----------------
#define BATCH   32
#define NIN     1024     // input capsules (N)
#define DIN     256      // input feature dim (K of GEMM)
#define NC      32       // num output capsules
#define DCAP    64       // dim per output capsule (GEMM N = 64 per capsule)
#define WCOLS   2048     // NC*DCAP

#define A_STRIDE  264    // 256 + 8 pad, in bf16 elems (row stride of A planes)
#define UH_STRIDE 72     // 64 + 8 pad, in f16 elems (row stride of u_hat)

// ---------------- LDS partition (bytes) ----------------
#define OFF_UHAT 0
#define SZ_UHAT  (NIN * UH_STRIDE * 2)          // 147456  (_Float16)
#define OFF_WHI  (OFF_UHAT + SZ_UHAT)
#define SZ_W     (4 * 8 * 32 * 16 * 2)          // 32768   (swizzled B frags, bf16)
#define OFF_WLO  (OFF_WHI + SZ_W)
#define OFF_AHI  (OFF_WLO + SZ_W)
#define SZ_A     (2 * 32 * A_STRIDE * 2)        // 33792   (double-buffered, bf16)
#define OFF_ALO  (OFF_AHI + SZ_A)
#define OFF_BL   (OFF_ALO + SZ_A)               // b logits, 1024 f32 = 4096
#define OFF_CE   (OFF_BL + 4096)                // exp(b - max), 1024 f32 = 4096
#define OFF_SR   (OFF_CE + 4096)                // reduction scratch, 256 f32 = 1024
#define OFF_VV   (OFF_SR + 1024)                // v vector, 64 f32 = 256
#define SMEM_TOTAL (OFF_VV + 256)               // = 290048 bytes < 320 KB

// ---------------- WMMA helper ----------------
__device__ __forceinline__ v8f wmma_bf16(v16bf a, v16bf b, v8f c) {
    // D = A(16x32 bf16) * B(32x16 bf16) + C(16x16 f32)
    return __builtin_amdgcn_wmma_f32_16x16x32_bf16(
        /*neg_a=*/false, a, /*neg_b=*/false, b,
        /*c_mod=*/(short)0, c, /*reuse_a=*/false, /*reuse_b=*/false);
}

// A-fragment gather: 8 bf16 at p[0..7] and 8 bf16 at p[16..23] (16B-aligned each)
__device__ __forceinline__ v16bf ldA16(const __bf16* p) {
    v8bf a = *(const v8bf*)p;
    v8bf b = *(const v8bf*)(p + 16);
    return __builtin_shufflevector(a, b, 0,1,2,3,4,5,6,7,8,9,10,11,12,13,14,15);
}

__device__ __forceinline__ float blockReduceMax(float v, float* sr, int tid) {
    sr[tid] = v; __syncthreads();
    for (int off = 128; off > 0; off >>= 1) {
        if (tid < off) sr[tid] = fmaxf(sr[tid], sr[tid + off]);
        __syncthreads();
    }
    float r = sr[0]; __syncthreads();
    return r;
}

__device__ __forceinline__ float blockReduceSum(float v, float* sr, int tid) {
    sr[tid] = v; __syncthreads();
    for (int off = 128; off > 0; off >>= 1) {
        if (tid < off) sr[tid] = sr[tid] + sr[tid + off];
        __syncthreads();
    }
    float r = sr[0]; __syncthreads();
    return r;
}

// One workgroup per (batch, capsule): GEMM tile [1024 x 64] into LDS (f16),
// then 3 dynamic-routing iterations entirely in LDS, write v[64] to global.
__global__ __launch_bounds__(256, 1)
void capsule_fused_kernel(const float* __restrict__ u,   // [32,1024,256]
                          const float* __restrict__ w,   // [256,2048]
                          float* __restrict__ out) {     // [32,32,64]
    extern __shared__ __align__(16) char smem[];
    _Float16* uhat = (_Float16*)(smem + OFF_UHAT);
    __bf16*   Whi  = (__bf16*)(smem + OFF_WHI);
    __bf16*   Wlo  = (__bf16*)(smem + OFF_WLO);
    __bf16*   Ahi  = (__bf16*)(smem + OFF_AHI);
    __bf16*   Alo  = (__bf16*)(smem + OFF_ALO);
    float*    b_l  = (float*)(smem + OFF_BL);
    float*    cexp = (float*)(smem + OFF_CE);
    float*    sred = (float*)(smem + OFF_SR);
    float*    vvec = (float*)(smem + OFF_VV);

    const int tid  = threadIdx.x;
    const int lane = tid & 31;
    const int wave = tid >> 5;
    const int bat  = blockIdx.x >> 5;   // batch index
    const int cap  = blockIdx.x & 31;   // output capsule index

    // ---- load W slice [256 x 64], split to hi/lo bf16, pre-swizzled into
    // ---- exact B-fragment order: frag(((nt*8+q)*32 + lane)*16 + e)
    {
        const int k = tid;                               // 256 threads = 256 k rows
        const float* wrow = w + (size_t)k * WCOLS + (size_t)cap * DCAP;
        const int q     = k >> 5;
        const int kl    = k & 31;
        const int e     = (kl & 7) | ((kl & 16) >> 1);
        const int lbase = (kl & 8) ? 16 : 0;
        #pragma unroll 4
        for (int n = 0; n < DCAP; ++n) {
            float  x  = wrow[n];
            __bf16 hi = (__bf16)x;
            __bf16 lo = (__bf16)(x - (float)hi);
            int pos = ((((n >> 4) * 8 + q) * 32) + lbase + (n & 15)) * 16 + e;
            Whi[pos] = hi;
            Wlo[pos] = lo;
        }
        for (int j = tid; j < NIN; j += 256) b_l[j] = 0.0f;  // routing logits
    }

    // ---- GEMM: u_hat[j][c] = sum_k u[bat][j][k] * W[k][cap*64+c] ----
    // 32 M-chunks of 32 rows; 8 waves -> one 16x16 tile each per chunk.
    const int nt = wave & 3;        // n-tile (16 cols)
    const int mt = wave >> 2;       // m-tile within chunk (16 rows)
    const int koff  = (lane & 16) ? 8 : 0;
    const int rbase = (lane & 16) ? 8 : 0;
    const int arow  = mt * 16 + (lane & 15);
    const int col   = nt * 16 + (lane & 15);

    // per-thread A staging assignment: 32 rows x 256 cols per chunk
    const int ldrow = tid >> 3;              // 0..31
    const int ldcol = (tid & 7) * 32;        // 8 threads x 32 floats per row
    const float* ubase = u + ((size_t)bat * NIN) * DIN;

    float4 stage[8];
    // prefetch chunk 0
    {
        const float* src = ubase + (size_t)ldrow * DIN + ldcol;
        #pragma unroll
        for (int t = 0; t < 8; ++t) stage[t] = ((const float4*)src)[t];
        __bf16* dh = Ahi + (size_t)ldrow * A_STRIDE + ldcol;
        __bf16* dl = Alo + (size_t)ldrow * A_STRIDE + ldcol;
        #pragma unroll
        for (int t = 0; t < 8; ++t) {
            float xs[4] = {stage[t].x, stage[t].y, stage[t].z, stage[t].w};
            #pragma unroll
            for (int s = 0; s < 4; ++s) {
                __bf16 hi = (__bf16)xs[s];
                dh[t * 4 + s] = hi;
                dl[t * 4 + s] = (__bf16)(xs[s] - (float)hi);
            }
        }
    }
    __syncthreads();

    // ---- hoist this wave's 16 B fragments (W slice) into registers: they are
    // ---- invariant across all 32 M-chunks (128 VGPRs; 2 waves/SIMD -> fits).
    v16bf Bh[8], Bl[8];
    #pragma unroll
    for (int q = 0; q < 8; ++q) {
        const int bpi = ((nt * 8 + q) * 32 + lane) * 16;
        Bh[q] = *(const v16bf*)(Whi + bpi);
        Bl[q] = *(const v16bf*)(Wlo + bpi);
    }

    for (int ch = 0; ch < 32; ++ch) {
        const int buf = ch & 1;
        const bool more = (ch + 1) < 32;
        // issue next chunk's global loads early (latency hidden behind WMMA)
        if (more) {
            const float* src = ubase + (size_t)((ch + 1) * 32 + ldrow) * DIN + ldcol;
            #pragma unroll
            for (int t = 0; t < 8; ++t) stage[t] = ((const float4*)src)[t];
        }

        // compute this wave's 16x16 tile over full K=256
        v8f acc = {0.f, 0.f, 0.f, 0.f, 0.f, 0.f, 0.f, 0.f};
        const __bf16* Ah = Ahi + (size_t)(buf * 32 + arow) * A_STRIDE;
        const __bf16* Al = Alo + (size_t)(buf * 32 + arow) * A_STRIDE;
        #pragma unroll
        for (int q = 0; q < 8; ++q) {
            const int k0 = q * 32 + koff;
            v16bf ah = ldA16(Ah + k0);
            v16bf al = ldA16(Al + k0);
            acc = wmma_bf16(ah, Bh[q], acc);   // hi*hi
            acc = wmma_bf16(al, Bh[q], acc);   // lo*hi
            acc = wmma_bf16(ah, Bl[q], acc);   // hi*lo
        }
        // C fragment -> u_hat (f16) in LDS
        const int gr = ch * 32 + mt * 16 + rbase;
        #pragma unroll
        for (int r = 0; r < 8; ++r)
            uhat[(size_t)(gr + r) * UH_STRIDE + col] = (_Float16)acc[r];

        // write-back next chunk's A planes (after compute, before barrier)
        if (more) {
            __bf16* dh = Ahi + (size_t)((buf ^ 1) * 32 + ldrow) * A_STRIDE + ldcol;
            __bf16* dl = Alo + (size_t)((buf ^ 1) * 32 + ldrow) * A_STRIDE + ldcol;
            #pragma unroll
            for (int t = 0; t < 8; ++t) {
                float xs[4] = {stage[t].x, stage[t].y, stage[t].z, stage[t].w};
                #pragma unroll
                for (int s = 0; s < 4; ++s) {
                    __bf16 hi = (__bf16)xs[s];
                    dh[t * 4 + s] = hi;
                    dl[t * 4 + s] = (__bf16)(xs[s] - (float)hi);
                }
            }
        }
        __syncthreads();
    }

    // ---- dynamic routing: 3 iterations, all in LDS ----
    const int c = tid & 63;
    const int g = tid >> 6;
    for (int it = 0; it < 3; ++it) {
        // softmax over the 1024 logits: max, exp, sum
        float mx = -3.402823466e38f;
        for (int j = tid; j < NIN; j += 256) mx = fmaxf(mx, b_l[j]);
        mx = blockReduceMax(mx, sred, tid);
        float se = 0.f;
        for (int j = tid; j < NIN; j += 256) {
            float e = __expf(b_l[j] - mx);
            cexp[j] = e;
            se += e;
        }
        se = blockReduceSum(se, sred, tid);
        const float inv = 1.0f / se;

        // s[c] = sum_j c_j * u_hat[j][c]   (4 partial groups over j)
        float part = 0.f;
        for (int j = g; j < NIN; j += 4)
            part += cexp[j] * (float)uhat[(size_t)j * UH_STRIDE + c];
        part *= inv;
        sred[tid] = part;
        __syncthreads();
        if (tid < 64)
            vvec[tid] = sred[tid] + sred[tid + 64] + sred[tid + 128] + sred[tid + 192];
        __syncthreads();
        // squash: v = s / sqrt(||s||^2 + eps)
        if (tid == 0) {
            float n2 = 0.f;
            #pragma unroll 8
            for (int cc = 0; cc < 64; ++cc) { float s = vvec[cc]; n2 += s * s; }
            sred[0] = 1.0f / sqrtf(n2 + 1e-7f);
        }
        __syncthreads();
        const float scale = sred[0];
        if (tid < 64) vvec[tid] *= scale;
        __syncthreads();

        if (it < 2) {
            // b_j += v . u_hat[j][:]   (vectorized v8h reads of u_hat rows)
            for (int j = tid; j < NIN; j += 256) {
                const _Float16* row = uhat + (size_t)j * UH_STRIDE;
                float dot = 0.f;
                #pragma unroll
                for (int cb = 0; cb < 8; ++cb) {
                    v8h hv = *(const v8h*)(row + cb * 8);
                    #pragma unroll
                    for (int s = 0; s < 8; ++s)
                        dot += vvec[cb * 8 + s] * (float)hv[s];
                }
                b_l[j] += dot;
            }
            __syncthreads();
        } else {
            if (tid < 64)
                out[(size_t)blockIdx.x * 64 + tid] = vvec[tid];
        }
    }
}

extern "C" void kernel_launch(void* const* d_in, const int* in_sizes, int n_in,
                              void* d_out, int out_size, void* d_ws, size_t ws_size,
                              hipStream_t stream) {
    const float* u = (const float*)d_in[0];   // u_vecs [32,1024,256]
    const float* w = (const float*)d_in[1];   // W [1,256,2048] -> [256,2048]
    float* out = (float*)d_out;               // v [32,32,64]
    (void)in_sizes; (void)n_in; (void)d_ws; (void)ws_size; (void)out_size;

    dim3 grid(BATCH * NC);   // 1024 workgroups, one per (batch, capsule)
    dim3 block(256);         // 8 wave32
    capsule_fused_kernel<<<grid, block, SMEM_TOTAL, stream>>>(u, w, out);
}
